// SESiLUTypePattern_23313082482876
// MI455X (gfx1250) — 8-bit matrix operand order repaired and checked by reading, compiled, not run on hardware
//
#include <hip/hip_runtime.h>
#include <math.h>


#define HW    16384
#define NB    64
#define C_IN  32
#define C_MID 8

typedef int   v8i __attribute__((ext_vector_type(8)));
typedef float v4f __attribute__((ext_vector_type(4)));

__device__ __forceinline__ int quanti(float x, float inv_s) {
  float q = fminf(fmaxf(x * inv_s, -127.f), 127.f);
  return (int)rintf(q);
}

__device__ __forceinline__ int pack4i(int a, int b, int c, int d) {
  unsigned lo = __builtin_amdgcn_perm((unsigned)b, (unsigned)a, 0x0c0c0400u);
  unsigned hi = __builtin_amdgcn_perm((unsigned)d, (unsigned)c, 0x0c0c0400u);
  return (int)__builtin_amdgcn_perm(hi, lo, 0x05040100u);
}

__device__ __forceinline__ float sigm(float z) {
  return __builtin_amdgcn_rcpf(1.f + __expf(-z));
}

__global__ void k0_init(const float* __restrict__ w_reduce,
                        const float* __restrict__ w_expand,
                        unsigned char* __restrict__ ws) {
  unsigned*    amax     = (unsigned*)ws;
  float*       scale_wr = (float*)(ws + 16);
  float*       scale_we = (float*)(ws + 64);
  signed char* wq_r     = (signed char*)(ws + 192);
  signed char* wq_e     = (signed char*)(ws + 448);
  int t = threadIdx.x;
  if (t < 2) amax[t] = 0u;
  if (t < 8) {
    const float* row = w_reduce + t * 32;
    float am = 1e-12f;
    for (int c = 0; c < 32; ++c) am = fmaxf(am, fabsf(row[c]));
    scale_wr[t] = am / 127.f;
    float inv = 127.f / am;
    for (int c = 0; c < 32; ++c) wq_r[t * 32 + c] = (signed char)quanti(row[c], inv);
  }
  if (t >= 8 && t < 40) {
    int o = t - 8;
    const float* row = w_expand + o * 8;
    float am = 1e-12f;
    for (int c = 0; c < 8; ++c) am = fmaxf(am, fabsf(row[c]));
    scale_we[o] = am / 127.f;
    float inv = 127.f / am;
    for (int c = 0; c < 8; ++c) wq_e[o * 8 + c] = (signed char)quanti(row[c], inv);
  }
}

__global__ __launch_bounds__(256)
void k1_amax_x(const v4f* __restrict__ x4, unsigned* __restrict__ amax) {
  const long n4 = (long)NB * C_IN * HW / 4;
  float m = 0.f;
  for (long i = (long)blockIdx.x * blockDim.x + threadIdx.x; i < n4;
       i += (long)gridDim.x * blockDim.x) {
    v4f v = x4[i];
    m = fmaxf(m, fmaxf(fmaxf(fabsf(v.x), fabsf(v.y)),
                       fmaxf(fabsf(v.z), fabsf(v.w))));
  }
  for (int off = 16; off; off >>= 1) m = fmaxf(m, __shfl_xor(m, off, 32));
  if ((threadIdx.x & 31) == 0) atomicMax(amax, __float_as_uint(m));
}

__global__ __launch_bounds__(256)
void k2_reduce_silu(const float* __restrict__ x,
                    const float* __restrict__ b_reduce,
                    unsigned char* __restrict__ ws) {
  const float* scale_wr = (const float*)(ws + 16);
  const signed char* wq_r = (const signed char*)(ws + 192);
  float* h = (float*)(ws + 1024);
  unsigned* amax_h = ((unsigned*)ws) + 1;

  float ax     = fmaxf(__uint_as_float(((const unsigned*)ws)[0]), 1e-12f);
  float sx     = ax / 127.f;
  float inv_sx = 127.f / ax;

  const int lane = threadIdx.x & 31;
  const int n    = lane & 15;
  const int half = lane >> 4;

  v8i Bm = {};
  if (n < 8) {
    int4 w = *(const int4*)(wq_r + n * 32 + half * 16);
    Bm[0] = w.x; Bm[1] = w.y; Bm[2] = w.z; Bm[3] = w.w;
  }
  const float ssw = ((n < 8) ? scale_wr[n] : 0.f) * sx;
  const float bo  = (n < 8) ? b_reduce[n] : 0.f;

  const int wave   = (int)((blockIdx.x * blockDim.x + threadIdx.x) >> 5);
  const int nwaves = (int)((gridDim.x * blockDim.x) >> 5);
  const int ntiles = NB * (HW / 16);
  float lmax = 0.f;

  for (int t = wave; t < ntiles; t += nwaves) {
    const int b   = t >> 10;
    const int hw0 = (t & 1023) << 4;
    const float* xb = x + (size_t)b * C_IN * HW + hw0 + n;

    v8i A = {};
#pragma unroll
    for (int g = 0; g < 2; ++g) {
      const int cbase = half * 16 + g * 8;
#pragma unroll
      for (int v = 0; v < 2; ++v) {
        const int c0 = cbase + v * 4;
        A[g * 2 + v] = pack4i(quanti(xb[(size_t)(c0 + 0) * HW], inv_sx),
                              quanti(xb[(size_t)(c0 + 1) * HW], inv_sx),
                              quanti(xb[(size_t)(c0 + 2) * HW], inv_sx),
                              quanti(xb[(size_t)(c0 + 3) * HW], inv_sx));
      }
    }

    v8i Cz = {};
    v8i D = __builtin_amdgcn_wmma_i32_16x16x64_iu8(
         true, A,  true, Bm, Cz,
         false,  false);

    if (n < 8) {
      float* hp = h + ((size_t)b * C_MID + n) * HW + hw0 + half * 8;
      float s[8];
#pragma unroll
      for (int j = 0; j < 8; ++j) {
        float z = (float)D[j] * ssw + bo;
        s[j] = z * sigm(z);
        lmax = fmaxf(lmax, fabsf(s[j]));
      }
      *(v4f*)(hp + 0) = (v4f){s[0], s[1], s[2], s[3]};
      *(v4f*)(hp + 4) = (v4f){s[4], s[5], s[6], s[7]};
    }
  }
  for (int off = 16; off; off >>= 1) lmax = fmaxf(lmax, __shfl_xor(lmax, off, 32));
  if (lane == 0) atomicMax(amax_h, __float_as_uint(lmax));
}

__global__ __launch_bounds__(256)
void k3_expand_out(const float* __restrict__ identity,
                   const float* __restrict__ b_expand,
                   const unsigned char* __restrict__ ws,
                   float* __restrict__ out) {
  const float* scale_we = (const float*)(ws + 64);
  const signed char* wq_e = (const signed char*)(ws + 448);
  const float* h = (const float*)(ws + 1024);

  float ah     = fmaxf(__uint_as_float(((const unsigned*)ws)[1]), 1e-12f);
  float sh     = ah / 127.f;
  float inv_sh = 127.f / ah;

  const int lane = threadIdx.x & 31;
  const int n    = lane & 15;
  const int half = lane >> 4;

  v8i B0 = {}, B1 = {};
  if (half == 0) {
    int2 w0 = *(const int2*)(wq_e + n * 8);
    int2 w1 = *(const int2*)(wq_e + (n + 16) * 8);
    B0[0] = w0.x; B0[1] = w0.y;
    B1[0] = w1.x; B1[1] = w1.y;
  }
  const float sws[2] = { scale_we[n] * sh, scale_we[n + 16] * sh };
  const float bos[2] = { b_expand[n],      b_expand[n + 16] };

  const int wave   = (int)((blockIdx.x * blockDim.x + threadIdx.x) >> 5);
  const int nwaves = (int)((gridDim.x * blockDim.x) >> 5);
  const int ntiles = NB * (HW / 16);

  for (int t = wave; t < ntiles; t += nwaves) {
    const int b   = t >> 10;
    const int hw0 = (t & 1023) << 4;

    v8i A = {};
    if (half == 0) {
      const float* hb = h + (size_t)b * C_MID * HW + hw0 + n;
      A[0] = pack4i(quanti(hb[0 * HW], inv_sh), quanti(hb[1 * HW], inv_sh),
                    quanti(hb[2 * HW], inv_sh), quanti(hb[3 * HW], inv_sh));
      A[1] = pack4i(quanti(hb[4 * HW], inv_sh), quanti(hb[5 * HW], inv_sh),
                    quanti(hb[6 * HW], inv_sh), quanti(hb[7 * HW], inv_sh));
    }

    v8i Cz = {};
    v8i D0 = __builtin_amdgcn_wmma_i32_16x16x64_iu8(true, A, true, B0, Cz, false, false);
    v8i D1 = __builtin_amdgcn_wmma_i32_16x16x64_iu8(true, A, true, B1, Cz, false, false);

#pragma unroll
    for (int T = 0; T < 2; ++T) {
      const int o = n + 16 * T;
      const size_t base = ((size_t)b * C_IN + o) * HW + hw0 + half * 8;
      v4f i0 = __builtin_nontemporal_load((const v4f*)(identity + base));
      v4f i1 = __builtin_nontemporal_load((const v4f*)(identity + base + 4));
      float r[8];
      const float ss = sws[T], bb = bos[T];
#pragma unroll
      for (int j = 0; j < 8; ++j) {
        int acc = (T == 0) ? D0[j] : D1[j];
        float g = (float)acc * ss + bb;
        float idv = (j < 4) ? i0[j] : i1[j - 4];
        r[j] = idv * sigm(g);
      }
      v4f r0 = {r[0], r[1], r[2], r[3]};
      v4f r1 = {r[4], r[5], r[6], r[7]};
      __builtin_nontemporal_store(r0, (v4f*)(out + base));
      __builtin_nontemporal_store(r1, (v4f*)(out + base + 4));
    }
  }
}

extern "C" void kernel_launch(void* const* d_in, const int* in_sizes, int n_in,
                              void* d_out, int out_size, void* d_ws, size_t ws_size,
                              hipStream_t stream) {
  const float* x        = (const float*)d_in[0];
  const float* identity = (const float*)d_in[1];
  const float* w_reduce = (const float*)d_in[2];
  const float* b_reduce = (const float*)d_in[3];
  const float* w_expand = (const float*)d_in[4];
  const float* b_expand = (const float*)d_in[5];
  float* out = (float*)d_out;
  unsigned char* ws = (unsigned char*)d_ws;

  k0_init<<<1, 64, 0, stream>>>(w_reduce, w_expand, ws);
  k1_amax_x<<<2048, 256, 0, stream>>>((const v4f*)x, (unsigned*)ws);
  k2_reduce_silu<<<2048, 256, 0, stream>>>(x, b_reduce, ws);
  k3_expand_out<<<2048, 256, 0, stream>>>(identity, b_expand, ws, out);
}
